// MultiHeadSelfAttention_20246475833975
// MI455X (gfx1250) — matrix operand order repaired and checked by reading, not compiled, not run on hardware
//
#include <hip/hip_runtime.h>
#include <hip/hip_bf16.h>

// ---------------------------------------------------------------------------
// MultiHeadSelfAttention (B=8, C=512, H=W=32, nh=8, hd=64, n=1024), fp32 io.
//
// MI455X plan:
//  * All GEMMs via v_wmma_f32_16x16x32_f16 (f16 in, f32 accumulate).
//  * Flash-attention fusion: the 256MB energy matrix never touches memory.
//  * Working set (~34MB f16 staging in d_ws) is L2-resident (192MB L2),
//    so loads go straight global->VGPR (no LDS staging hop needed).
//  * Register blocking: projection 32x32/wave (12 WMMA per 8 loads),
//    attention 32 q-rows/wave (16 WMMA per 8 B-loads), to keep the XDL
//    pipe fed instead of being B128-load bound.
//  * P-transpose through LDS uses an interleaved key permutation so the
//    D-layout -> A-layout bounce is 8 packed b32 DS stores, not 16 b16.
// ---------------------------------------------------------------------------

typedef _Float16 v16h __attribute__((ext_vector_type(16)));
typedef _Float16 h8   __attribute__((ext_vector_type(8)));
typedef _Float16 h2   __attribute__((ext_vector_type(2)));
typedef float    v8f  __attribute__((ext_vector_type(8)));

#define NB   8
#define NC   512
#define NPIX 1024   // 32*32
#define NH   8
#define HD   64

__device__ __forceinline__ v16h make16(h8 lo, h8 hi) {
  v16h r;
#pragma unroll
  for (int i = 0; i < 8; ++i) { r[i] = lo[i]; r[i + 8] = hi[i]; }
  return r;
}

// A-matrix 16x32 f16 (MxK), row-major source, leading dim ld (halves).
// ISA layout: lane (r = lane&15, s = lane>>4): elements 0..7 = K 8s+0..7,
// elements 8..15 = K 16+8s+0..7, all for row M=r.
__device__ __forceinline__ v16h load_a16x32(const _Float16* base, int ld) {
  const int lane = threadIdx.x;
  const int r = lane & 15, s = lane >> 4;
  const _Float16* p = base + r * ld + 8 * s;
  h8 lo = *(const h8*)(p);
  h8 hi = *(const h8*)(p + 16);
  return make16(lo, hi);
}

// B-matrix 32x16 f16 (KxN), row-major source, leading dim ld (halves).
// ISA layout: lane (n = lane&15, s = lane>>4): elements 0..7 = K 8s+0..7,
// elements 8..15 = K 16+8s+0..7, all for column N=n.
__device__ __forceinline__ v16h load_b32x16(const _Float16* base, int ld) {
  const int lane = threadIdx.x;
  const _Float16* p = base + (lane & 15);
  const int ko = 8 * (lane >> 4);
  v16h r;
#pragma unroll
  for (int i = 0; i < 8; ++i) { r[i] = p[(ko + i) * ld]; r[i + 8] = p[(16 + ko + i) * ld]; }
  return r;
}

// B-tile load from a per-lane column pointer (V, keys permuted: slot k -> key (k&1)?16+(k>>1):k>>1).
__device__ __forceinline__ v16h load_b_row(const _Float16* rowp) {
  const int ko = 8 * ((int)threadIdx.x >> 4);
  v16h r;
#pragma unroll
  for (int i = 0; i < 16; ++i) {
    const int k = (i < 8) ? (ko + i) : (16 + ko + i - 8);
    const int key = (k & 1) ? (16 + (k >> 1)) : (k >> 1);
    r[i] = rowp[key * HD];
  }
  return r;
}

__device__ __forceinline__ v8f wmma_f16(v16h a, v16h b, v8f c) {
  return __builtin_amdgcn_wmma_f32_16x16x32_f16(
      false, a, false, b, (short)0, c, false, false);
}

// Row-wise reductions over the 16 lanes holding one C/D row (N dimension).
__device__ __forceinline__ float rmax16(float v) {
#pragma unroll
  for (int m = 1; m < 16; m <<= 1) v = fmaxf(v, __shfl_xor(v, m, 16));
  return v;
}
__device__ __forceinline__ float rsum16(float v) {
#pragma unroll
  for (int m = 1; m < 16; m <<= 1) v += __shfl_xor(v, m, 16);
  return v;
}

// ---------------------------------------------------------------------------
// Conversion kernels: f16 staging copies with WMMA-friendly layouts.
// ---------------------------------------------------------------------------

// x [b][c][p] f32 -> Xt [b][p][c] f16 (A operand of the projection GEMM).
__global__ void cvt_x_kernel(const float* __restrict__ x,
                             _Float16* __restrict__ Xt) {
  int idx = blockIdx.x * blockDim.x + threadIdx.x;   // over NB*NC*NPIX
  int p = idx & (NPIX - 1);
  int c = (idx >> 10) & (NC - 1);
  int b = idx >> 19;
  Xt[(((b << 10) + p) << 9) + c] = (_Float16)x[idx];
}

// W [o][c] f32 -> Wt [c][o] f16 (B operand). blockIdx.y picks q/k/v.
__global__ void cvt_w_kernel(const float* __restrict__ Wq,
                             const float* __restrict__ Wk,
                             const float* __restrict__ Wv,
                             _Float16* __restrict__ Wqt,
                             _Float16* __restrict__ Wkt,
                             _Float16* __restrict__ Wvt) {
  int idx = blockIdx.x * blockDim.x + threadIdx.x;   // over NC*NC
  int o = idx >> 9;
  int c = idx & (NC - 1);
  const float* src = (blockIdx.y == 0) ? Wq : (blockIdx.y == 1) ? Wk : Wv;
  _Float16* dst    = (blockIdx.y == 0) ? Wqt : (blockIdx.y == 1) ? Wkt : Wvt;
  dst[(c << 9) + o] = (_Float16)src[(o << 9) + c];
}

// ---------------------------------------------------------------------------
// Fused QKV projection, 32x32 per-wave tile: Out[p,o] = Xt[p,:].Wt[:,o]+bias.
// Per K-step: 2 A loads + 6 B loads feed 12 WMMAs (A reused across q/k/v
// and across both o-tiles; B reused across both p-tiles).
//   Q,V stored f16 as [bh][p][d] ; K stored f16 as [bh][d][p].
// ---------------------------------------------------------------------------
__global__ __launch_bounds__(128)
void qkv_proj_kernel(const _Float16* __restrict__ Xt,
                     const _Float16* __restrict__ Wqt,
                     const _Float16* __restrict__ Wkt,
                     const _Float16* __restrict__ Wvt,
                     const float* __restrict__ bq,
                     const float* __restrict__ bk,
                     const float* __restrict__ bv,
                     _Float16* __restrict__ Qh,
                     _Float16* __restrict__ Kh,
                     _Float16* __restrict__ Vh) {
  const int lane = threadIdx.x;
  const int s = lane >> 4, col = lane & 15;
  const int o0 = (blockIdx.x * 4 + threadIdx.y) * 32;   // 32 output channels
  const int p0 = blockIdx.y * 32;                       // 32 pixels
  const int b  = blockIdx.z;

  const _Float16* xa = Xt + (size_t)(((b << 10) + p0) << 9);
  const _Float16* wsrc[3] = {Wqt, Wkt, Wvt};

  v8f acc[3][2][2];   // [matrix][p-tile][o-tile]
#pragma unroll
  for (int w = 0; w < 3; ++w)
#pragma unroll
    for (int mt = 0; mt < 2; ++mt)
#pragma unroll
      for (int nt = 0; nt < 2; ++nt) acc[w][mt][nt] = (v8f){};

  for (int c0 = 0; c0 < NC; c0 += 32) {
    v16h a0 = load_a16x32(xa + c0, NC);
    v16h a1 = load_a16x32(xa + (16 * NC) + c0, NC);
#pragma unroll
    for (int w = 0; w < 3; ++w) {
      v16h bt;
      bt = load_b32x16(wsrc[w] + (c0 << 9) + o0, NC);
      acc[w][0][0] = wmma_f16(a0, bt, acc[w][0][0]);
      acc[w][1][0] = wmma_f16(a1, bt, acc[w][1][0]);
      bt = load_b32x16(wsrc[w] + (c0 << 9) + o0 + 16, NC);
      acc[w][0][1] = wmma_f16(a0, bt, acc[w][0][1]);
      acc[w][1][1] = wmma_f16(a1, bt, acc[w][1][1]);
    }
  }

  // Epilogue: bias add, f16 convert, scatter to Q/K/V layouts.
#pragma unroll
  for (int nt = 0; nt < 2; ++nt) {
    const int o = o0 + nt * 16 + col;
    const int h = o >> 6, d = o & 63;
    const int bh = b * NH + h;
    const float bqv = bq[o], bkv = bk[o], bvv = bv[o];
#pragma unroll
    for (int mt = 0; mt < 2; ++mt) {
#pragma unroll
      for (int i = 0; i < 8; ++i) {
        const int p = p0 + mt * 16 + i + 8 * s;
        Qh[(((bh << 10) + p) << 6) + d] = (_Float16)(acc[0][mt][nt][i] + bqv);
        Kh[(((bh << 6) + d) << 10) + p] = (_Float16)(acc[1][mt][nt][i] + bkv);
        Vh[(((bh << 10) + p) << 6) + d] = (_Float16)(acc[2][mt][nt][i] + bvv);
      }
    }
  }
}

// ---------------------------------------------------------------------------
// Flash attention + epilogue (gamma*out + x), 32 query rows per wave.
// Per 32-key step: 4 K B-tiles feed 8 logit WMMAs (both q-tiles), online
// softmax in f32, P bounced D-layout -> A-layout through LDS with packed
// b32 stores (interleaved key permutation), then 4 V B-tiles feed 8 PV WMMAs.
// Key permutation: WMMA-K slot 2c <-> key c, slot 2c+1 <-> key 16+c;
// V row addressing applies pi(k) = (k&1) ? 16+(k>>1) : k>>1 to each K slot k.
// ---------------------------------------------------------------------------
__global__ __launch_bounds__(128)
void attn_kernel(const _Float16* __restrict__ Qh,
                 const _Float16* __restrict__ Kh,
                 const _Float16* __restrict__ Vh,
                 const float* __restrict__ x,
                 const float* __restrict__ gamma,
                 float* __restrict__ out) {
  __shared__ _Float16 Pbuf[4][2][16][32];   // 2KB per wave

  const int lane = threadIdx.x, wv = threadIdx.y;
  const int s = lane >> 4, col = lane & 15;
  const int p0 = (blockIdx.x * 4 + wv) * 32;      // 32 query rows
  const int h = blockIdx.y, b = blockIdx.z;
  const int bh = b * NH + h;
  const int pi = (lane & 1) ? (16 + (lane >> 1)) : (lane >> 1); // V row perm

  const _Float16* Qb = Qh + ((size_t)bh << 16);   // [p][d], ld=64
  const _Float16* Kb = Kh + ((size_t)bh << 16);   // [d][p], ld=1024
  const _Float16* Vb = Vh + ((size_t)bh << 16);   // [p][d], ld=64

  // Q tiles held in registers for the whole key sweep.
  v16h aq[2][2];
#pragma unroll
  for (int t = 0; t < 2; ++t) {
    aq[t][0] = load_a16x32(Qb + (p0 + 16 * t) * HD, HD);
    aq[t][1] = load_a16x32(Qb + (p0 + 16 * t) * HD + 32, HD);
  }

  float m[2][8], l[2][8];
  v8f oacc[2][4];
#pragma unroll
  for (int t = 0; t < 2; ++t) {
#pragma unroll
    for (int i = 0; i < 8; ++i) { m[t][i] = -3.0e38f; l[t][i] = 0.f; }
#pragma unroll
    for (int n = 0; n < 4; ++n) oacc[t][n] = (v8f){};
  }

  for (int k0 = 0; k0 < NPIX; k0 += 32) {
    if (k0 + 32 < NPIX) {                         // hint next K/V tiles
      __builtin_prefetch(Kb + (k0 + 32), 0, 3);
      __builtin_prefetch(Vb + (k0 + 32) * HD, 0, 3);
    }

    // K B-tiles, shared by both q-tiles (8 logit WMMAs from 4 loads).
    const v16h bk00 = load_b32x16(Kb + k0, NPIX);
    const v16h bk01 = load_b32x16(Kb + (32 << 10) + k0, NPIX);
    const v16h bk10 = load_b32x16(Kb + k0 + 16, NPIX);
    const v16h bk11 = load_b32x16(Kb + (32 << 10) + k0 + 16, NPIX);

#pragma unroll
    for (int t = 0; t < 2; ++t) {
      v8f sa = (v8f){}, sb = (v8f){};
      sa = wmma_f16(aq[t][0], bk00, sa);
      sa = wmma_f16(aq[t][1], bk01, sa);
      sb = wmma_f16(aq[t][0], bk10, sb);
      sb = wmma_f16(aq[t][1], bk11, sb);

      // Online softmax: element i is row (i + 8*s), col = lane&15.
#pragma unroll
      for (int i = 0; i < 8; ++i) {
        float mv = rmax16(fmaxf(sa[i], sb[i]));
        float mn = fmaxf(m[t][i], mv);
        float sc = __expf(m[t][i] - mn);
        float pa = __expf(sa[i] - mn);   // key rel = col      -> K slot 2*col
        float pb = __expf(sb[i] - mn);   // key rel = 16+col   -> K slot 2*col+1
        float rs = rsum16(pa + pb);
        l[t][i] = l[t][i] * sc + rs;
        m[t][i] = mn;
        oacc[t][0][i] *= sc; oacc[t][1][i] *= sc;
        oacc[t][2][i] *= sc; oacc[t][3][i] *= sc;
        h2 pk; pk[0] = (_Float16)pa; pk[1] = (_Float16)pb;
        ((h2*)&Pbuf[wv][t][i + 8 * s][0])[col] = pk;   // one b32 DS store
      }
    }

    // P tiles back in A layout (same wave: DS in order, compiler waits).
    const v16h ap0 = load_a16x32(&Pbuf[wv][0][0][0], 32);
    const v16h ap1 = load_a16x32(&Pbuf[wv][1][0][0], 32);

    // V B-tiles (permuted rows), shared by both q-tiles: 8 WMMAs / 4 loads.
    const _Float16* vrow = Vb + k0 * HD + col;
#pragma unroll
    for (int n = 0; n < 4; ++n) {
      const v16h bvt = load_b_row(vrow + n * 16);
      oacc[0][n] = wmma_f16(ap0, bvt, oacc[0][n]);
      oacc[1][n] = wmma_f16(ap1, bvt, oacc[1][n]);
    }
  }

  // Epilogue: out[b, h*64+d, p] = gamma * (O/l) + x[...]
  const float g = gamma[0];
#pragma unroll
  for (int t = 0; t < 2; ++t) {
#pragma unroll
    for (int i = 0; i < 8; ++i) {
      const float inv = (l[t][i] > 0.f) ? (1.0f / l[t][i]) : 0.f;
      const int p = p0 + t * 16 + i + 8 * s;
#pragma unroll
      for (int n = 0; n < 4; ++n) {
        const int d = n * 16 + col;
        const int idx = (((b << 9) + h * HD + d) << 10) + p;
        out[idx] = g * (oacc[t][n][i] * inv) + x[idx];
      }
    }
  }
}

// ---------------------------------------------------------------------------
extern "C" void kernel_launch(void* const* d_in, const int* in_sizes, int n_in,
                              void* d_out, int out_size, void* d_ws, size_t ws_size,
                              hipStream_t stream) {
  const float* x     = (const float*)d_in[0];
  const float* Wq    = (const float*)d_in[1];
  const float* bq    = (const float*)d_in[2];
  const float* Wk    = (const float*)d_in[3];
  const float* bk    = (const float*)d_in[4];
  const float* Wv    = (const float*)d_in[5];
  const float* bv    = (const float*)d_in[6];
  const float* gamma = (const float*)d_in[7];
  float* out = (float*)d_out;

  // Workspace carve-up (f16 staging, ~33.5 MB total; all 16B-aligned).
  char* ws = (char*)d_ws;
  _Float16* Xt  = (_Float16*)ws; ws += (size_t)NB * NPIX * NC * 2;      // 8 MB
  _Float16* Wqt = (_Float16*)ws; ws += (size_t)NC * NC * 2;             // 0.5 MB
  _Float16* Wkt = (_Float16*)ws; ws += (size_t)NC * NC * 2;
  _Float16* Wvt = (_Float16*)ws; ws += (size_t)NC * NC * 2;
  _Float16* Qh  = (_Float16*)ws; ws += (size_t)NB * NH * NPIX * HD * 2; // 8 MB
  _Float16* Kh  = (_Float16*)ws; ws += (size_t)NB * NH * NPIX * HD * 2; // 8 MB
  _Float16* Vh  = (_Float16*)ws; ws += (size_t)NB * NH * NPIX * HD * 2; // 8 MB

  // 1) f32 -> f16 staging with WMMA-friendly layouts.
  cvt_x_kernel<<<(NB * NC * NPIX) / 256, 256, 0, stream>>>(x, Xt);
  cvt_w_kernel<<<dim3((NC * NC) / 256, 3), 256, 0, stream>>>(
      Wq, Wk, Wv, Wqt, Wkt, Wvt);

  // 2) Fused QKV projection (32x32 per-wave tiles, shared A across q/k/v).
  qkv_proj_kernel<<<dim3(NC / 128, NPIX / 32, NB), dim3(32, 4), 0, stream>>>(
      Xt, Wqt, Wkt, Wvt, bq, bk, bv, Qh, Kh, Vh);

  // 3) Flash attention (32 q-rows per wave) + gamma*out + x epilogue.
  attn_kernel<<<dim3(NPIX / 128, NH, NB), dim3(32, 4), 0, stream>>>(
      Qh, Kh, Vh, x, gamma, out);
}
